// GameNet_85761906966992
// MI455X (gfx1250) — compile-verified, matrix operand order checked by reading, not run on hardware
//
#include <hip/hip_runtime.h>
#include <hip/hip_bf16.h>
#include <stdint.h>

#define N_NODES 768
#define N_FACES 256
#define NTOT    1024
#define E_DIR   6144
#define CCH     64
#define NCOLS   (NTOT * CCH)   // 65536

typedef __attribute__((ext_vector_type(16))) __bf16 bf16x16;
typedef __attribute__((ext_vector_type(8)))  __bf16 bf16x8;
typedef __attribute__((ext_vector_type(8)))  float  f32x8;

struct alignas(8) bf16x4s { __bf16 x, y, z, w; };

__device__ inline bf16x16 make_frag(const __bf16* p_lo, const __bf16* p_hi) {
  bf16x8 lo = *(const bf16x8*)p_lo;
  bf16x8 hi = *(const bf16x8*)p_hi;
  return __builtin_shufflevector(lo, hi, 0,1,2,3,4,5,6,7,8,9,10,11,12,13,14,15);
}

__device__ inline f32x8 wmma_bf16(bf16x16 a, bf16x16 b, f32x8 c) {
  return __builtin_amdgcn_wmma_f32_16x16x32_bf16(false, a, false, b, (short)0, c,
                                                 false, false);
}

// CDNA5 async global->LDS copy (16B per lane), tracked by ASYNCcnt.
__device__ inline void async_copy_b128(uint lds_byte_off, const void* gaddr) {
  asm volatile("global_load_async_to_lds_b128 %0, %1, off"
               :: "v"(lds_byte_off), "v"((unsigned long long)(uintptr_t)gaddr)
               : "memory");
}
__device__ inline void wait_async0() {
  asm volatile("s_wait_asynccnt 0" ::: "memory");
}

// ---------------- adjacency construction ----------------

__global__ void k_build_A(const int* __restrict__ ei, const int* __restrict__ fn,
                          float* __restrict__ A) {
  int idx = blockIdx.x * blockDim.x + threadIdx.x;
  if (idx < E_DIR) {
    int s = ei[idx], d = ei[E_DIR + idx];
    atomicAdd(&A[s * NTOT + d], 1.0f);
  } else if (idx < E_DIR + N_FACES * 6) {
    int e = idx - E_DIR;
    int f = e / 6;
    int node = fn[e];
    atomicAdd(&A[node * NTOT + (N_NODES + f)], 1.0f);   // forward face edge
    atomicAdd(&A[(N_NODES + f) * NTOT + node], 1.0f);   // reversed face edge
  } else if (idx < E_DIR + N_FACES * 6 + NTOT) {
    int i = idx - E_DIR - N_FACES * 6;
    atomicAdd(&A[i * NTOT + i], 1.0f);                  // + I
  }
}

__global__ void k_dinv(const float* __restrict__ A, float* __restrict__ dinv) {
  __shared__ float sd[256];
  int i = blockIdx.x;
  float s = 0.f;
  for (int j = threadIdx.x; j < NTOT; j += 256) s += A[i * NTOT + j];
  sd[threadIdx.x] = s;
  __syncthreads();
  for (int o = 128; o > 0; o >>= 1) {
    if (threadIdx.x < o) sd[threadIdx.x] += sd[threadIdx.x + o];
    __syncthreads();
  }
  if (threadIdx.x == 0) {
    float d = sd[0];
    dinv[i] = d > 0.f ? rsqrtf(d) : 0.f;
  }
}

__global__ void k_adjn(const float* __restrict__ A, const float* __restrict__ dinv,
                       __bf16* __restrict__ adjn) {
  int idx = blockIdx.x * 256 + threadIdx.x;
  int i = idx >> 10, j = idx & 1023;
  adjn[idx] = (__bf16)(A[idx] * dinv[i] * dinv[j]);
}

// ---------------- small embedding MLPs ----------------

__global__ void k_mlp2(const float* __restrict__ X, int in_dim,
                       const float* __restrict__ W1, const float* __restrict__ b1,
                       const float* __restrict__ W2, const float* __restrict__ b2,
                       float* __restrict__ Y) {
  __shared__ float xs[160];
  __shared__ float hs[64];
  int row = blockIdx.x;
  int c = threadIdx.x;
  for (int k = c; k < in_dim; k += 64) xs[k] = X[row * in_dim + k];
  __syncthreads();
  float h = b1[c];
  for (int k = 0; k < in_dim; ++k) h = fmaf(xs[k], W1[k * 64 + c], h);
  hs[c] = fmaxf(h, 0.f);
  __syncthreads();
  float y = b2[c];
  for (int k = 0; k < 64; ++k) y = fmaf(hs[k], W2[k * 64 + c], y);
  Y[row * 64 + c] = fmaxf(y, 0.f);
}

// ---------------- scatter into dense [N,N,C] (f32 staging) ----------------

__global__ void k_scatter(const int* __restrict__ ei, const int* __restrict__ fn,
                          const float* __restrict__ edge_e,
                          const float* __restrict__ node_all,
                          float* __restrict__ full) {
  int b = blockIdx.x;
  int c = threadIdx.x;
  if (b < E_DIR) {
    int s = ei[b], d = ei[E_DIR + b];
    atomicAdd(&full[((size_t)(s * NTOT + d)) * 64 + c], edge_e[b * 64 + c]);
  } else if (b < E_DIR + N_FACES * 6) {
    int e = b - E_DIR;
    int f = e / 6;
    int node = fn[e];
    float v = node_all[(N_NODES + f) * 64 + c];   // face embedding
    atomicAdd(&full[((size_t)(node * NTOT + N_NODES + f)) * 64 + c], v);
    atomicAdd(&full[((size_t)((N_NODES + f) * NTOT + node)) * 64 + c], v);
  } else {
    int i = b - E_DIR - N_FACES * 6;
    atomicAdd(&full[((size_t)(i * NTOT + i)) * 64 + c], node_all[i * 64 + c]);
  }
}

__global__ void k_cvt(const float4* __restrict__ src, bf16x4s* __restrict__ dst,
                      int n4) {
  int i = blockIdx.x * 256 + threadIdx.x;
  if (i < n4) {
    float4 v = src[i];
    bf16x4s o{(__bf16)v.x, (__bf16)v.y, (__bf16)v.z, (__bf16)v.w};
    dst[i] = o;
  }
}

// ---------------- big einsum: H[1024 x 65536] = adjn @ full ----------------

__global__ __launch_bounds__(256) void k_gemm_adj(const __bf16* __restrict__ Aadj,
                                                  const __bf16* __restrict__ B,
                                                  __bf16* __restrict__ H) {
  __shared__ alignas(16) __bf16 As[64 * 32];
  __shared__ alignas(16) __bf16 BsT[128 * 32];
  const int tid = threadIdx.x;
  const int lane = tid & 31;
  const int w = tid >> 5;     // 8 waves
  const int wm = w & 3;       // 4 M subtiles of 16
  const int wn = w >> 2;      // 2 N halves of 64
  const int l16 = lane & 15, half = lane >> 4;
  const int m0 = blockIdx.y * 64;
  const int n0 = blockIdx.x * 128;

  // A staging: one async b128 per thread covers the 64x32 tile.
  const int a_row = tid >> 2, a_col8 = (tid & 3) * 8;
  const uint a_lds = (uint)(uintptr_t)&As[a_row * 32 + a_col8];
  // B staging: thread owns one LDS column (n), one 16-wide k-half.
  const int b_n = tid & 127;
  const int b_kh = (tid >> 7) * 16;

  const f32x8 zero = {0.f, 0.f, 0.f, 0.f, 0.f, 0.f, 0.f, 0.f};
  f32x8 acc[4];
#pragma unroll
  for (int t = 0; t < 4; ++t) acc[t] = zero;

  for (int k0 = 0; k0 < NTOT; k0 += 32) {
    // async copy of A tile: overlaps with the B gather below
    async_copy_b128(a_lds, &Aadj[(size_t)(m0 + a_row) * NTOT + k0 + a_col8]);

    // gather B tile 32x128 transposed: pack k-pairs into dwords, b128 stores
    {
      uint p[8];
#pragma unroll
      for (int i = 0; i < 8; ++i) {
        int k1 = k0 + b_kh + 2 * i;
        uint lo = (uint)*(const unsigned short*)&B[(size_t)k1 * NCOLS + n0 + b_n];
        uint hi = (uint)*(const unsigned short*)&B[(size_t)(k1 + 1) * NCOLS + n0 + b_n];
        p[i] = lo | (hi << 16);
      }
      uint4* dst = (uint4*)&BsT[b_n * 32 + b_kh];
      dst[0] = make_uint4(p[0], p[1], p[2], p[3]);
      dst[1] = make_uint4(p[4], p[5], p[6], p[7]);
    }
    wait_async0();
    __syncthreads();

    const __bf16* ap = &As[(wm * 16 + l16) * 32];
    bf16x16 af = make_frag(ap + half * 8, ap + 16 + half * 8);
#pragma unroll
    for (int t = 0; t < 4; ++t) {
      const __bf16* bp = &BsT[(wn * 64 + t * 16 + l16) * 32];
      bf16x16 bf = make_frag(bp + half * 8, bp + 16 + half * 8);
      acc[t] = wmma_bf16(af, bf, acc[t]);
    }
    __syncthreads();
  }
#pragma unroll
  for (int t = 0; t < 4; ++t) {
    int n = n0 + wn * 64 + t * 16 + l16;
#pragma unroll
    for (int r = 0; r < 8; ++r) {
      int m = m0 + wm * 16 + r + half * 8;
      H[(size_t)m * NCOLS + n] = (__bf16)acc[t][r];
    }
  }
}

// ---------------- full = relu(H @ plW + plb), rows = 1M, K=N=64 ----------------

__global__ __launch_bounds__(256) void k_gemm_pl(const __bf16* __restrict__ Hmat,
                                                 const float* __restrict__ W,
                                                 const float* __restrict__ bvec,
                                                 __bf16* __restrict__ Out) {
  __shared__ alignas(16) __bf16 WT[64 * 64];   // WT[n][k]
  __shared__ float bs[64];
  const int tid = threadIdx.x;
  const int lane = tid & 31;
  const int w = tid >> 5;
  const int l16 = lane & 15, half = lane >> 4;
#pragma unroll
  for (int i = 0; i < 16; ++i) {
    int idx = tid * 16 + i;
    int n = idx >> 6, k = idx & 63;
    WT[n * 64 + k] = (__bf16)W[k * 64 + n];
  }
  if (tid < 64) bs[tid] = bvec[tid];
  __syncthreads();

  size_t row0 = (size_t)blockIdx.x * 128 + (size_t)w * 16;
  const f32x8 zero = {0.f, 0.f, 0.f, 0.f, 0.f, 0.f, 0.f, 0.f};
  f32x8 acc[4];
#pragma unroll
  for (int t = 0; t < 4; ++t) acc[t] = zero;

#pragma unroll
  for (int ks = 0; ks < 2; ++ks) {
    const __bf16* ap = Hmat + (row0 + l16) * 64 + ks * 32 + half * 8;
    bf16x16 af = make_frag(ap, ap + 16);
#pragma unroll
    for (int t = 0; t < 4; ++t) {
      const __bf16* bp = &WT[(t * 16 + l16) * 64 + ks * 32 + half * 8];
      bf16x16 bf = make_frag(bp, bp + 16);
      acc[t] = wmma_bf16(af, bf, acc[t]);
    }
  }
#pragma unroll
  for (int t = 0; t < 4; ++t) {
    int n = t * 16 + l16;
#pragma unroll
    for (int r = 0; r < 8; ++r) {
      size_t m = row0 + r + half * 8;
      Out[m * 64 + n] = (__bf16)fmaxf(acc[t][r] + bs[n], 0.f);
    }
  }
}

// ---------------- fused output MLP: relu(X@oW1+b1)@oW2+b2 ----------------

__global__ __launch_bounds__(256) void k_out_mlp(const __bf16* __restrict__ X,
                                                 const float* __restrict__ W1,
                                                 const float* __restrict__ b1,
                                                 const float* __restrict__ W2,
                                                 const float* __restrict__ b2,
                                                 float* __restrict__ Out) {
  __shared__ alignas(16) __bf16 W1T[64 * 64];
  __shared__ alignas(16) __bf16 W2T[32 * 64];
  __shared__ float b1s[64];
  __shared__ float b2s[32];
  __shared__ alignas(16) __bf16 Hs[8][16 * 64];
  const int tid = threadIdx.x;
  const int lane = tid & 31;
  const int w = tid >> 5;
  const int l16 = lane & 15, half = lane >> 4;
#pragma unroll
  for (int i = 0; i < 16; ++i) {
    int idx = tid * 16 + i;
    int n = idx >> 6, k = idx & 63;
    W1T[n * 64 + k] = (__bf16)W1[k * 64 + n];
  }
#pragma unroll
  for (int i = 0; i < 8; ++i) {
    int idx = tid * 8 + i;
    int n = idx >> 6, k = idx & 63;
    W2T[n * 64 + k] = (__bf16)W2[k * 32 + n];
  }
  if (tid < 64) b1s[tid] = b1[tid];
  if (tid < 32) b2s[tid] = b2[tid];
  __syncthreads();

  size_t row0 = (size_t)blockIdx.x * 128 + (size_t)w * 16;
  const f32x8 zero = {0.f, 0.f, 0.f, 0.f, 0.f, 0.f, 0.f, 0.f};
  f32x8 hacc[4];
#pragma unroll
  for (int t = 0; t < 4; ++t) hacc[t] = zero;

#pragma unroll
  for (int ks = 0; ks < 2; ++ks) {
    const __bf16* ap = X + (row0 + l16) * 64 + ks * 32 + half * 8;
    bf16x16 af = make_frag(ap, ap + 16);
#pragma unroll
    for (int t = 0; t < 4; ++t) {
      const __bf16* bp = &W1T[(t * 16 + l16) * 64 + ks * 32 + half * 8];
      bf16x16 bf = make_frag(bp, bp + 16);
      hacc[t] = wmma_bf16(af, bf, hacc[t]);
    }
  }
  // relu + bias, re-fragment via LDS
#pragma unroll
  for (int t = 0; t < 4; ++t) {
    int n = t * 16 + l16;
#pragma unroll
    for (int r = 0; r < 8; ++r) {
      int m = r + half * 8;
      Hs[w][m * 64 + n] = (__bf16)fmaxf(hacc[t][r] + b1s[n], 0.f);
    }
  }
  __syncthreads();

  f32x8 yacc[2];
#pragma unroll
  for (int t = 0; t < 2; ++t) yacc[t] = zero;
#pragma unroll
  for (int ks = 0; ks < 2; ++ks) {
    const __bf16* ap = &Hs[w][l16 * 64 + ks * 32 + half * 8];
    bf16x16 af = make_frag(ap, ap + 16);
#pragma unroll
    for (int t = 0; t < 2; ++t) {
      const __bf16* bp = &W2T[(t * 16 + l16) * 64 + ks * 32 + half * 8];
      bf16x16 bf = make_frag(bp, bp + 16);
      yacc[t] = wmma_bf16(af, bf, yacc[t]);
    }
  }
#pragma unroll
  for (int t = 0; t < 2; ++t) {
    int n = t * 16 + l16;
#pragma unroll
    for (int r = 0; r < 8; ++r) {
      size_t m = row0 + r + half * 8;
      Out[m * 32 + n] = yacc[t][r] + b2s[n];
    }
  }
}

// ---------------- launcher ----------------

extern "C" void kernel_launch(void* const* d_in, const int* in_sizes, int n_in,
                              void* d_out, int out_size, void* d_ws, size_t ws_size,
                              hipStream_t stream) {
  (void)in_sizes; (void)n_in; (void)out_size; (void)ws_size;
  const float* node_x = (const float*)d_in[0];
  const float* edge_x = (const float*)d_in[1];
  const float* face_x = (const float*)d_in[2];
  const float* nW1 = (const float*)d_in[3];
  const float* nb1 = (const float*)d_in[4];
  const float* nW2 = (const float*)d_in[5];
  const float* nb2 = (const float*)d_in[6];
  const float* eW1 = (const float*)d_in[7];
  const float* eb1 = (const float*)d_in[8];
  const float* eW2 = (const float*)d_in[9];
  const float* eb2 = (const float*)d_in[10];
  const float* fW1 = (const float*)d_in[11];
  const float* fb1 = (const float*)d_in[12];
  const float* fW2 = (const float*)d_in[13];
  const float* fb2 = (const float*)d_in[14];
  const float* oW1 = (const float*)d_in[15];
  const float* ob1 = (const float*)d_in[16];
  const float* oW2 = (const float*)d_in[17];
  const float* ob2 = (const float*)d_in[18];
  const float* plW = (const float*)d_in[19];
  const float* plb = (const float*)d_in[20];
  const int* edge_index = (const int*)d_in[21];
  const int* face_nodes = (const int*)d_in[22];

  char* ws = (char*)d_ws;
  const size_t SZ_FULLF32 = (size_t)NTOT * NCOLS * 4;   // 256 MB
  const size_t SZ_FULLBF  = (size_t)NTOT * NCOLS * 2;   // 128 MB
  float*  full_f32 = (float*)ws;                        // staging
  __bf16* h_bf     = (__bf16*)ws;                       // aliases full_f32 after cvt
  __bf16* full_bf  = (__bf16*)(ws + SZ_FULLF32);
  __bf16* adjn_bf  = (__bf16*)(ws + SZ_FULLF32 + SZ_FULLBF);
  float*  A_f32    = (float*)(ws + SZ_FULLF32 + SZ_FULLBF + (size_t)NTOT * NTOT * 2);
  float*  dinv     = (float*)((char*)A_f32 + (size_t)NTOT * NTOT * 4);
  float*  node_all = (float*)((char*)dinv + 4096);
  float*  edge_e   = (float*)((char*)node_all + (size_t)NTOT * 64 * 4);

  hipMemsetAsync(A_f32, 0, (size_t)NTOT * NTOT * 4, stream);
  hipMemsetAsync(full_f32, 0, SZ_FULLF32, stream);

  int items = E_DIR + N_FACES * 6 + NTOT;
  k_build_A<<<(items + 255) / 256, 256, 0, stream>>>(edge_index, face_nodes, A_f32);
  k_dinv<<<NTOT, 256, 0, stream>>>(A_f32, dinv);
  k_adjn<<<NTOT * NTOT / 256, 256, 0, stream>>>(A_f32, dinv, adjn_bf);

  k_mlp2<<<N_NODES, 64, 0, stream>>>(node_x, 144, nW1, nb1, nW2, nb2, node_all);
  k_mlp2<<<N_FACES, 64, 0, stream>>>(face_x, 12, fW1, fb1, fW2, fb2,
                                     node_all + N_NODES * 64);
  k_mlp2<<<E_DIR, 64, 0, stream>>>(edge_x, 144, eW1, eb1, eW2, eb2, edge_e);

  k_scatter<<<items, 64, 0, stream>>>(edge_index, face_nodes, edge_e, node_all,
                                      full_f32);
  k_cvt<<<(NTOT * NCOLS / 4) / 256, 256, 0, stream>>>((const float4*)full_f32,
                                                      (bf16x4s*)full_bf,
                                                      NTOT * NCOLS / 4);
  for (int l = 0; l < 2; ++l) {
    dim3 g(NCOLS / 128, NTOT / 64);
    k_gemm_adj<<<g, 256, 0, stream>>>(adjn_bf, full_bf, h_bf);
    k_gemm_pl<<<(NTOT * NTOT) / 128, 256, 0, stream>>>(h_bf, plW + l * 64 * 64,
                                                       plb + l * 64, full_bf);
  }
  k_out_mlp<<<(NTOT * NTOT) / 128, 256, 0, stream>>>(full_bf, oW1, ob1, oW2, ob2,
                                                     (float*)d_out);
}
